// CLOSpreadModel_18133351923780
// MI455X (gfx1250) — matrix operand order repaired and checked by reading, not compiled, not run on hardware
//
#include <hip/hip_runtime.h>
#include <hip/hip_bf16.h>

typedef __attribute__((ext_vector_type(16))) _Float16 v16h;
typedef __attribute__((ext_vector_type(8)))  float    v8f;

static constexpr int KK   = 64;  // knots per component
static constexpr int TILE = 16;  // samples per WMMA tile
static constexpr int NB   = 16;  // buckets

// ---- A-matrix (16x32 f16) lane layout: lane holds row m=lane&15.
// element e (0..15) -> K = kb + e, kb = (lane>=16 ? 16 : 0).
__device__ __forceinline__ v16h load_knots_A(const float* __restrict__ kn, int chunk, int kb) {
  v16h r;
#pragma unroll
  for (int e = 0; e < 16; ++e) {
    int k = chunk * 32 + kb + e;
    r[e] = (_Float16)kn[k];
  }
  return r;
}

// ---- B-matrix (32x16 f16) lane layout: lane holds column n=lane&15,
// element e -> K = kb + e, kb = (lane>=16 ? 16 : 0).
__device__ __forceinline__ v16h load_w_B(const float* __restrict__ w, int chunk, int kb) {
  v16h r;
#pragma unroll
  for (int e = 0; e < 16; ++e) r[e] = (_Float16)w[chunk * 32 + kb + e];
  return r;
}

__device__ __forceinline__ v16h relu_feat(_Float16 x, v16h kn) {
  v16h a;
#pragma unroll
  for (int e = 0; e < 16; ++e) {
    _Float16 d = x - kn[e];           // -> v_pk_add_f16 (neg) + v_pk_max_num_f16
    a[e] = d > (_Float16)0 ? d : (_Float16)0;
  }
  return a;
}

__global__ __launch_bounds__(256) void hinge_wmma_kernel(
    const float* __restrict__ mvoc, const int* __restrict__ bucket_idx,
    const float* __restrict__ lev, const float* __restrict__ wapx, const float* __restrict__ cpnx,
    const float* __restrict__ base_knots, const float* __restrict__ base_w, const float* __restrict__ base_b,
    const float* __restrict__ adj_knots, const float* __restrict__ adj_w, const float* __restrict__ adj_b,
    const float* __restrict__ idx_knots, const float* __restrict__ idx_w, const float* __restrict__ idx_b,
    const float* __restrict__ wap_knots, const float* __restrict__ wap_w, const float* __restrict__ wap_b,
    const float* __restrict__ cpn_knots, const float* __restrict__ cpn_w, const float* __restrict__ cpn_b,
    const float* __restrict__ bias,
    float* __restrict__ out, int n_tiles)
{
  // Per-bucket adj params pre-converted to f16 in WMMA lane layouts.
  // linear index = b*64 + half*32 + chunk*16 + e   (2 KB per table)
  __shared__ __align__(32) _Float16 sKA[NB * 64];  // A-side knots
  __shared__ __align__(32) _Float16 sWB[NB * 64];  // B-side weights

  for (int i = threadIdx.x; i < NB * 64; i += blockDim.x) {
    const int e = i & 15, j = (i >> 4) & 1, half = (i >> 5) & 1, b = i >> 6;
    sKA[i] = (_Float16)adj_knots[b * KK + j * 32 + half * 16 + e];
    sWB[i] = (_Float16)adj_w    [b * KK + j * 32 + half * 16 + e];
  }
  __syncthreads();

  const int lane = threadIdx.x & 31;
  const int half = (lane >= 16) ? 1 : 0;
  const int kbA  = half * 16;   // A layout K base
  const int kbB  = half * 16;   // B layout K base
  const int row  = lane & 15;   // both A's m and B's n for this lane

  const int waves_per_block = blockDim.x >> 5;
  const int wave = blockIdx.x * waves_per_block + (threadIdx.x >> 5);
  const int n_waves = gridDim.x * waves_per_block;

  // Hoist shared-component knots (A layout) and weights (B layout) into regs.
  const float* kp[4] = {base_knots, idx_knots, wap_knots, cpn_knots};
  const float* wp[4] = {base_w, idx_w, wap_w, cpn_w};
  v16h kA[4][2], wB[4][2];
#pragma unroll
  for (int c = 0; c < 4; ++c)
#pragma unroll
    for (int j = 0; j < 2; ++j) {
      kA[c][j] = load_knots_A(kp[c], j, kbA);
      wB[c][j] = load_w_B(wp[c], j, kbB);
    }

  const float sb = base_b[0] + idx_b[0] + wap_b[0] + cpn_b[0] + bias[0];

  for (int t = wave; t < n_tiles; t += n_waves) {
    const int s = t * TILE + row;
    _Float16 xs[4];
    xs[0] = (_Float16)mvoc[s];
    xs[1] = (_Float16)lev[s];
    xs[2] = (_Float16)wapx[s];
    xs[3] = (_Float16)cpnx[s];
    const int b = bucket_idx[s];

    // adj operands from LDS: 2 x v16h each = ds_load_b128 x4 per table
    const v16h* pak = (const v16h*)&sKA[b * 64 + half * 32];
    const v16h* paw = (const v16h*)&sWB[b * 64 + half * 32];
    v16h ak0 = pak[0], ak1 = pak[1];
    v16h aw0 = paw[0], aw1 = paw[1];

    // Two independent accumulation chains (5 WMMAs each) to shorten the
    // serial WMMA RAW dependency; summed at diagonal extraction.
    v8f acc0 = {}, acc1 = {};
#pragma unroll
    for (int c = 0; c < 2; ++c)
#pragma unroll
      for (int j = 0; j < 2; ++j)
        acc0 = __builtin_amdgcn_wmma_f32_16x16x32_f16(
            false, relu_feat(xs[c], kA[c][j]), false, wB[c][j], (short)0, acc0, false, false);
#pragma unroll
    for (int c = 2; c < 4; ++c)
#pragma unroll
      for (int j = 0; j < 2; ++j)
        acc1 = __builtin_amdgcn_wmma_f32_16x16x32_f16(
            false, relu_feat(xs[c], kA[c][j]), false, wB[c][j], (short)0, acc1, false, false);

    // Bucket-routed adjustment (diagonal trick): A rows use bucket(m) knots,
    // B cols use bucket(n) weights -> D[m,m] is sample m's adjustment.
    acc0 = __builtin_amdgcn_wmma_f32_16x16x32_f16(
        false, relu_feat(xs[0], ak0), false, aw0, (short)0, acc0, false, false);
    acc1 = __builtin_amdgcn_wmma_f32_16x16x32_f16(
        false, relu_feat(xs[0], ak1), false, aw1, (short)0, acc1, false, false);

    // Extract diagonal: m<8 at (lane=r, vgpr r); m>=8 at (lane=r+24, vgpr r).
    float res = 0.f;
#pragma unroll
    for (int r = 0; r < 8; ++r)
      if (lane == r || lane == r + 24) res = acc0[r] + acc1[r];

    if (lane < 8 || lane >= 24) {
      const int m = (lane < 8) ? lane : lane - 16;
      out[t * TILE + m] = res + sb + adj_b[b];  // this lane's b == bucket(m)
    }
  }
}

// Scalar tail kernel for N % 16 != 0 (WMMA path requires full 16-sample tiles).
__global__ void hinge_tail_kernel(
    const float* __restrict__ mvoc, const int* __restrict__ bucket_idx,
    const float* __restrict__ lev, const float* __restrict__ wapx, const float* __restrict__ cpnx,
    const float* __restrict__ base_knots, const float* __restrict__ base_w, const float* __restrict__ base_b,
    const float* __restrict__ adj_knots, const float* __restrict__ adj_w, const float* __restrict__ adj_b,
    const float* __restrict__ idx_knots, const float* __restrict__ idx_w, const float* __restrict__ idx_b,
    const float* __restrict__ wap_knots, const float* __restrict__ wap_w, const float* __restrict__ wap_b,
    const float* __restrict__ cpn_knots, const float* __restrict__ cpn_w, const float* __restrict__ cpn_b,
    const float* __restrict__ bias,
    float* __restrict__ out, int start, int n)
{
  int i = start + blockIdx.x * blockDim.x + threadIdx.x;
  if (i >= n) return;
  float acc = bias[0] + base_b[0] + idx_b[0] + wap_b[0] + cpn_b[0];
  const float xm = mvoc[i], xi = lev[i], xw = wapx[i], xc = cpnx[i];
  const int b = bucket_idx[i];
  acc += adj_b[b];
#pragma unroll 4
  for (int k = 0; k < KK; ++k) {
    acc += fmaxf(xm - base_knots[k], 0.f) * base_w[k];
    acc += fmaxf(xm - adj_knots[b * KK + k], 0.f) * adj_w[b * KK + k];
    acc += fmaxf(xi - idx_knots[k], 0.f) * idx_w[k];
    acc += fmaxf(xw - wap_knots[k], 0.f) * wap_w[k];
    acc += fmaxf(xc - cpn_knots[k], 0.f) * cpn_w[k];
  }
  out[i] = acc;
}

extern "C" void kernel_launch(void* const* d_in, const int* in_sizes, int n_in,
                              void* d_out, int out_size, void* d_ws, size_t ws_size,
                              hipStream_t stream) {
  const float* mvoc       = (const float*)d_in[0];
  const int*   bucket_idx = (const int*)  d_in[1];
  const float* lev        = (const float*)d_in[2];
  const float* wapx       = (const float*)d_in[3];
  const float* cpnx       = (const float*)d_in[4];
  const float* base_knots = (const float*)d_in[5];
  const float* base_w     = (const float*)d_in[6];
  const float* base_b     = (const float*)d_in[7];
  const float* adj_knots  = (const float*)d_in[8];
  const float* adj_w      = (const float*)d_in[9];
  const float* adj_b      = (const float*)d_in[10];
  const float* idx_knots  = (const float*)d_in[11];
  const float* idx_w      = (const float*)d_in[12];
  const float* idx_b      = (const float*)d_in[13];
  const float* wap_knots  = (const float*)d_in[14];
  const float* wap_w      = (const float*)d_in[15];
  const float* wap_b      = (const float*)d_in[16];
  const float* cpn_knots  = (const float*)d_in[17];
  const float* cpn_w      = (const float*)d_in[18];
  const float* cpn_b      = (const float*)d_in[19];
  const float* bias       = (const float*)d_in[20];
  float* out = (float*)d_out;

  const int n = in_sizes[0];
  const int n_tiles = n / TILE;
  const int rem = n % TILE;

  if (n_tiles > 0) {
    const int threads = 256;               // 8 wave32s
    const int waves_per_block = threads / 32;
    int blocks = (n_tiles + waves_per_block - 1) / waves_per_block;
    if (blocks > 512) blocks = 512;        // grid-stride over tiles
    hipLaunchKernelGGL(hinge_wmma_kernel, dim3(blocks), dim3(threads), 0, stream,
        mvoc, bucket_idx, lev, wapx, cpnx,
        base_knots, base_w, base_b,
        adj_knots, adj_w, adj_b,
        idx_knots, idx_w, idx_b,
        wap_knots, wap_w, wap_b,
        cpn_knots, cpn_w, cpn_b,
        bias, out, n_tiles);
  }
  if (rem > 0) {
    hipLaunchKernelGGL(hinge_tail_kernel, dim3(1), dim3(64), 0, stream,
        mvoc, bucket_idx, lev, wapx, cpnx,
        base_knots, base_w, base_b,
        adj_knots, adj_w, adj_b,
        idx_knots, idx_w, idx_b,
        wap_knots, wap_w, wap_b,
        cpn_knots, cpn_w, cpn_b,
        bias, out, n_tiles * TILE, n);
  }
}
